// E3Conv_16887811408323
// MI455X (gfx1250) — matrix operand order repaired and checked by reading, not compiled, not run on hardware
//
#include <hip/hip_runtime.h>
#include <hip/hip_bf16.h>
#include <math.h>

#define N_   16384
#define ER_  524288
#define EB_  32768
#define EE_  (ER_ + EB_)      /* 557056 edges, divisible by 256 and 16 */
#define CS_  64
#define CV_  32
#define EA_  16
#define INV_NORM 0.17149858514250882f   /* 1/sqrt((E_R+E_B)/N) = 1/sqrt(34) */
#define TILE_H 512                       /* halfs per packed B tile: 32 lanes x 16 */

typedef __attribute__((ext_vector_type(16))) _Float16 v16h;
typedef __attribute__((ext_vector_type(8)))  float    v8f;

// ---------------------------------------------------------------------------
// WMMA helpers (CDNA5 wave32, V_WMMA_F32_16X16X32_F16)
// ---------------------------------------------------------------------------
__device__ __forceinline__ v8f wmma16(v16h a, v16h b, v8f c) {
  return __builtin_amdgcn_wmma_f32_16x16x32_f16(
      /*neg_a=*/false, a, /*neg_b=*/false, b,
      /*c_mod=*/(short)0, c, /*reuse_a=*/false, /*reuse_b=*/false);
}

__device__ __forceinline__ v8f zero8() {
  v8f z;
#pragma unroll
  for (int i = 0; i < 8; ++i) z[i] = 0.f;
  return z;
}

// A: 16x32 f16 tile, M rows from f32 row-major matrix (K multiple of 32).
// Lanes 0-15: row M=lane, K {0..7,16..23}; lanes 16-31: K {8..15,24..31}.
__device__ __forceinline__ v16h load_A_f32(const float* M, int ld, int row0,
                                           int k0, int lane) {
  int row = row0 + (lane & 15);
  int kb  = k0 + ((lane >> 4) << 3);
  const float* rp = M + (size_t)row * ld;
  v16h a;
#pragma unroll
  for (int i = 0; i < 8; ++i) {
    a[i]     = (_Float16)rp[kb + i];
    a[8 + i] = (_Float16)rp[kb + 16 + i];
  }
  return a;
}

// A tile from fp16 edge attributes, actual K=16 zero-padded to 32.
__device__ __forceinline__ v16h load_A_edge(const _Float16* eah, int e0, int lane) {
  int row = e0 + (lane & 15);
  int kb  = (lane >> 4) << 3;                 // 0 or 8
  const _Float16* rp = eah + (size_t)row * EA_ + kb;
  v16h a;
#pragma unroll
  for (int i = 0; i < 8; ++i) {
    a[i]     = rp[i];                         // K = kb+i (< 16, valid)
    a[8 + i] = (_Float16)0.f;                 // K = kb+16+i (>= 16, pad)
  }
  return a;
}

// Branchless B: 32x16 f16 tile from f32 row-major W[K][ncols]; lane->N, halfs->K (same K order as A).
__device__ __forceinline__ v16h load_B_f32(const float* W, int ncols, int k0,
                                           int kmax, int n0, int lane, float scale) {
  int n = n0 + (lane & 15);
  int kb = k0 + ((lane >> 4) << 3);
  v16h b;
#pragma unroll
  for (int h = 0; h < 16; ++h) {
    int k = kb + h + (h >= 8 ? 8 : 0);
    float scl = (k < kmax) ? scale : 0.f;       // v_cndmask, no EXEC branch
    int kc = (k < kmax) ? k : 0;
    b[h] = (_Float16)(W[(size_t)kc * ncols + n] * scl);
  }
  return b;
}

// Pre-packed B tile: one aligned 32-byte vector load, zero VALU.
__device__ __forceinline__ v16h load_B_packed(const _Float16* P, int tile, int lane) {
  return *(const v16h*)(P + ((size_t)tile * 32 + lane) * 16);
}

// D: lane = column n0+(lane&15); VGPR r = row (lane>=16 ? 8 : 0)+r.
__device__ __forceinline__ void store_D(float* O, int ld, int row0, int n0,
                                        int lane, v8f d) {
  int n  = n0 + (lane & 15);
  int mb = row0 + ((lane >> 4) << 3);
#pragma unroll
  for (int r = 0; r < 8; ++r) O[(size_t)(mb + r) * ld + n] = d[r];
}

__device__ __forceinline__ void atomAddF(float* p, float v) {
  __hip_atomic_fetch_add(p, v, __ATOMIC_RELAXED, __HIP_MEMORY_SCOPE_AGENT);
}

__device__ __forceinline__ float sigm(float x) { return 1.f / (1.f + __expf(-x)); }

// ---------------------------------------------------------------------------
// Kernels
// ---------------------------------------------------------------------------
__global__ void k_zero(float* p, int n) {
  int i = blockIdx.x * 256 + threadIdx.x;
  if (i < n) p[i] = 0.f;
}

// Pre-pack an edge weight (nmats x 16 x ncols, f32) into WMMA B lane layout:
// dst[(mat*ntiles + tile)*32 + lane][h] = W[mat][k(lane,h)][tile*16+(lane&15)] * scale,
// k = 8*(lane>>4)+h for h<8; halfs 8..15 (K>=16) zero-filled (K padding to 32).
__global__ void k_packB(const float* __restrict__ W, int ncols, int ntiles,
                        int nmats, float scale, _Float16* __restrict__ dst) {
  int idx = blockIdx.x * 128 + threadIdx.x;
  int lane = idx & 31;
  int t = (idx >> 5) % ntiles;
  int m = (idx >> 5) / ntiles;
  if (m >= nmats) return;
  int kb = (lane >> 4) << 3;
  _Float16* dp = dst + ((size_t)(m * ntiles + t) * 32 + lane) * 16;
  const float* rp = W + ((size_t)m * EA_ + kb) * ncols + t * 16 + (lane & 15);
#pragma unroll
  for (int h = 0; h < 8; ++h) dp[h] = (_Float16)(rp[(size_t)h * ncols] * scale);
#pragma unroll
  for (int h = 8; h < 16; ++h) dp[h] = (_Float16)0.f;
}

// Per-edge geometry: src/dst, Y1 = sqrt(3)*vec/d, edge_attr (fp16, 16 ch).
__global__ void k_edge_geom(const float* __restrict__ pos,
                            const int* __restrict__ rad, const int* __restrict__ bnd,
                            const float* __restrict__ bond_tab,
                            int* __restrict__ srcA, int* __restrict__ dstA,
                            float* __restrict__ Y1, _Float16* __restrict__ eah) {
  int e = blockIdx.x * 256 + threadIdx.x;
  if (e >= EE_) return;
  int s, d, bm;
  if (e < ER_) { s = rad[e]; d = rad[ER_ + e]; bm = 0; }
  else { int eb = e - ER_; s = bnd[eb]; d = bnd[EB_ + eb]; bm = 1; }
  srcA[e] = s; dstA[e] = d;
  float dx = pos[3 * s] - pos[3 * d];
  float dy = pos[3 * s + 1] - pos[3 * d + 1];
  float dz = pos[3 * s + 2] - pos[3 * d + 2];
  float dist = sqrtf(dx * dx + dy * dy + dz * dz + 1e-12f);
  float inv = 1.7320508075688772f / dist;
  Y1[3 * e] = dx * inv; Y1[3 * e + 1] = dy * inv; Y1[3 * e + 2] = dz * inv;
  _Float16* ep = eah + (size_t)e * EA_;
#pragma unroll
  for (int j = 0; j < 8; ++j) ep[j] = (_Float16)bond_tab[bm * 8 + j];
#pragma unroll
  for (int i = 0; i < 8; ++i) {
    float t = 6.f * dist - (float)(i + 1);   // (d - center_i)/step, step=1/6
    ep[8 + i] = (_Float16)(__expf(-t * t) * 1.12f);
  }
}

// s[n][c] = atom_tab[type[n]][c] * (1 + c_noise*ns0_w[c])
__global__ void k_embed(const float* __restrict__ atom_tab, const int* __restrict__ types,
                        const float* __restrict__ ns0_w, const float* __restrict__ c_noise,
                        float* __restrict__ sO) {
  int idx = blockIdx.x * 256 + threadIdx.x;   // N*CS threads
  int n = idx >> 6, c = idx & 63;
  sO[idx] = atom_tab[types[n] * CS_ + c] * (1.f + c_noise[0] * ns0_w[c]);
}

// Layer-0 edge conv: per wave = 16 edges. w0/w1 via WMMA, fused atomic scatter.
__global__ void k_conv0(const _Float16* __restrict__ eah, const int* __restrict__ srcA,
                        const int* __restrict__ dstA, const float* __restrict__ Y1,
                        const float* __restrict__ sF,
                        const _Float16* __restrict__ pWe0, const _Float16* __restrict__ pWe1,
                        float* __restrict__ a_s, float* __restrict__ a_v) {
  int wave = (blockIdx.x * 256 + threadIdx.x) >> 5;
  int lane = threadIdx.x & 31;
  int e0 = wave * 16;
  __builtin_prefetch(eah + (size_t)(e0 + 16) * EA_, 0, 1);
  v16h A = load_A_edge(eah, e0, lane);
  int mb = (lane >> 4) << 3;
  int sn[8], dn[8]; float y0[8], y1v[8], y2[8];
#pragma unroll
  for (int r = 0; r < 8; ++r) {
    int e = e0 + mb + r;
    sn[r] = srcA[e]; dn[r] = dstA[e];
    y0[r] = Y1[3 * e]; y1v[r] = Y1[3 * e + 1]; y2[r] = Y1[3 * e + 2];
  }
#pragma unroll
  for (int ct = 0; ct < 4; ++ct) {
    v8f d0 = wmma16(A, load_B_packed(pWe0, ct, lane), zero8());
    v8f d1 = wmma16(A, load_B_packed(pWe1, ct, lane), zero8());
    int c = ct * 16 + (lane & 15);
#pragma unroll
    for (int r = 0; r < 8; ++r) {
      float sv = sF[sn[r] * CS_ + c];
      atomAddF(&a_s[(size_t)dn[r] * CS_ + c], d0[r] * sv);
      float wv = d1[r] * sv;
      atomAddF(&a_v[((size_t)0 * N_ + dn[r]) * CS_ + c], wv * y0[r]);
      atomAddF(&a_v[((size_t)1 * N_ + dn[r]) * CS_ + c], wv * y1v[r]);
      atomAddF(&a_v[((size_t)2 * N_ + dn[r]) * CS_ + c], wv * y2[r]);
    }
  }
}

// s_h = a_s @ Ws0 + s @ Wself0   (wave = 16-node x 16-col tile, 4 chained WMMAs)
__global__ void k_node0_s(const float* __restrict__ a_s, const float* __restrict__ sF,
                          const float* __restrict__ Ws0, const float* __restrict__ Wself0,
                          float* __restrict__ s_h) {
  int wave = (blockIdx.x * 256 + threadIdx.x) >> 5;
  int lane = threadIdx.x & 31;
  int tile = wave >> 2, ct = wave & 3;
  int n0 = tile * 16, cn0 = ct * 16;
  v8f acc = zero8();
#pragma unroll
  for (int kt = 0; kt < 2; ++kt)
    acc = wmma16(load_A_f32(a_s, CS_, n0, kt * 32, lane),
                 load_B_f32(Ws0, CS_, kt * 32, CS_, cn0, lane, 1.f), acc);
#pragma unroll
  for (int kt = 0; kt < 2; ++kt)
    acc = wmma16(load_A_f32(sF, CS_, n0, kt * 32, lane),
                 load_B_f32(Wself0, CS_, kt * 32, CS_, cn0, lane, 1.f), acc);
  store_D(s_h, CS_, n0, cn0, lane, acc);
}

// v_h[i] = a_v[i] @ Wv0 per plane
__global__ void k_node0_v(const float* __restrict__ a_v, const float* __restrict__ Wv0,
                          float* __restrict__ v_h) {
  int wave = (blockIdx.x * 256 + threadIdx.x) >> 5;
  int lane = threadIdx.x & 31;
  int tile = wave / 6, rem = wave % 6, pl = rem >> 1, ct = rem & 1;
  int n0 = tile * 16, cn0 = ct * 16;
  const float* Ab = a_v + (size_t)pl * N_ * CS_;
  v8f acc = zero8();
#pragma unroll
  for (int kt = 0; kt < 2; ++kt)
    acc = wmma16(load_A_f32(Ab, CS_, n0, kt * 32, lane),
                 load_B_f32(Wv0, CV_, kt * 32, CS_, cn0, lane, 1.f), acc);
  store_D(v_h + (size_t)pl * N_ * CV_, CV_, n0, cn0, lane, acc);
}

// ss/vv = (s_h, v_h) * (1 + c_noise*ns_w[l])
__global__ void k_scale(const float* __restrict__ s_h, const float* __restrict__ v_h,
                        const float* __restrict__ ns_w_l, const float* __restrict__ c_noise,
                        float* __restrict__ ss, float* __restrict__ vv) {
  int idx = blockIdx.x * 256 + threadIdx.x;   // N*96 threads
  int n = idx / 96, c = idx % 96;
  float sc = 1.f + c_noise[0] * ns_w_l[c];
  if (c < CS_) {
    ss[(size_t)n * CS_ + c] = s_h[(size_t)n * CS_ + c] * sc;
  } else {
    int d = c - CS_;
#pragma unroll
    for (int i = 0; i < 3; ++i) {
      size_t id = ((size_t)i * N_ + n) * CV_ + d;
      vv[id] = v_h[id] * sc;
    }
  }
}

// Layer edge conv: 5 edge-weight WMMAs (packed B) + fused gather/atomic scatter.
__global__ void k_conv(const _Float16* __restrict__ eah, const int* __restrict__ srcA,
                       const int* __restrict__ dstA, const float* __restrict__ Y1,
                       const float* __restrict__ ss, const float* __restrict__ vv,
                       const _Float16* __restrict__ pWss, const _Float16* __restrict__ pWvs,
                       const _Float16* __restrict__ pWsv, const _Float16* __restrict__ pWvv,
                       const _Float16* __restrict__ pWvx,
                       float* __restrict__ a_s, float* __restrict__ a_v) {
  int wave = (blockIdx.x * 256 + threadIdx.x) >> 5;
  int lane = threadIdx.x & 31;
  int e0 = wave * 16;
  __builtin_prefetch(eah + (size_t)(e0 + 16) * EA_, 0, 1);
  v16h A = load_A_edge(eah, e0, lane);
  int mb = (lane >> 4) << 3;
  int sn[8], dn[8]; float y0[8], y1v[8], y2[8];
#pragma unroll
  for (int r = 0; r < 8; ++r) {
    int e = e0 + mb + r;
    sn[r] = srcA[e]; dn[r] = dstA[e];
    y0[r] = Y1[3 * e]; y1v[r] = Y1[3 * e + 1]; y2[r] = Y1[3 * e + 2];
  }
  // scalar-channel part: m_s[:64] and m_v block 0
#pragma unroll
  for (int ct = 0; ct < 4; ++ct) {
    v8f dss = wmma16(A, load_B_packed(pWss, ct, lane), zero8());
    v8f dsv = wmma16(A, load_B_packed(pWsv, ct, lane), zero8());
    int c = ct * 16 + (lane & 15);
#pragma unroll
    for (int r = 0; r < 8; ++r) {
      float sv = ss[(size_t)sn[r] * CS_ + c];
      atomAddF(&a_s[(size_t)dn[r] * 96 + c], dss[r] * sv);
      float t = dsv[r] * sv;
      atomAddF(&a_v[((size_t)0 * N_ + dn[r]) * 128 + c], t * y0[r]);
      atomAddF(&a_v[((size_t)1 * N_ + dn[r]) * 128 + c], t * y1v[r]);
      atomAddF(&a_v[((size_t)2 * N_ + dn[r]) * 128 + c], t * y2[r]);
    }
  }
  // vector-channel part: m_s[64:96] (vdot) and m_v blocks 1 (vv) + 2 (cross)
#pragma unroll
  for (int ct = 0; ct < 2; ++ct) {
    v8f dvs = wmma16(A, load_B_packed(pWvs, ct, lane), zero8());
    v8f dvv = wmma16(A, load_B_packed(pWvv, ct, lane), zero8());
    v8f dvx = wmma16(A, load_B_packed(pWvx, ct, lane), zero8());
    int c = ct * 16 + (lane & 15);
#pragma unroll
    for (int r = 0; r < 8; ++r) {
      float v0 = vv[((size_t)0 * N_ + sn[r]) * CV_ + c];
      float v1 = vv[((size_t)1 * N_ + sn[r]) * CV_ + c];
      float v2 = vv[((size_t)2 * N_ + sn[r]) * CV_ + c];
      float vd = v0 * y0[r] + v1 * y1v[r] + v2 * y2[r];
      atomAddF(&a_s[(size_t)dn[r] * 96 + 64 + c], dvs[r] * vd);
      atomAddF(&a_v[((size_t)0 * N_ + dn[r]) * 128 + 64 + c], dvv[r] * v0);
      atomAddF(&a_v[((size_t)1 * N_ + dn[r]) * 128 + 64 + c], dvv[r] * v1);
      atomAddF(&a_v[((size_t)2 * N_ + dn[r]) * 128 + 64 + c], dvv[r] * v2);
      float cx0 = v1 * y2[r] - v2 * y1v[r];
      float cx1 = v2 * y0[r] - v0 * y2[r];
      float cx2 = v0 * y1v[r] - v1 * y0[r];
      atomAddF(&a_v[((size_t)0 * N_ + dn[r]) * 128 + 96 + c], dvx[r] * cx0);
      atomAddF(&a_v[((size_t)1 * N_ + dn[r]) * 128 + 96 + c], dvx[r] * cx1);
      atomAddF(&a_v[((size_t)2 * N_ + dn[r]) * 128 + 96 + c], dvx[r] * cx2);
    }
  }
}

// new_s = a_s@Wmix_s + ss@Wself_s; fused skip-gate into s_h
__global__ void k_node_s(const float* __restrict__ a_s, const float* __restrict__ ss,
                         const float* __restrict__ Wmix, const float* __restrict__ Wself,
                         const float* __restrict__ skw, const float* __restrict__ skb,
                         const float* __restrict__ c_noise, float* __restrict__ s_h) {
  int wave = (blockIdx.x * 256 + threadIdx.x) >> 5;
  int lane = threadIdx.x & 31;
  int tile = wave >> 2, ct = wave & 3;
  int n0 = tile * 16, cn0 = ct * 16;
  v8f acc = zero8();
#pragma unroll
  for (int kt = 0; kt < 3; ++kt)
    acc = wmma16(load_A_f32(a_s, 96, n0, kt * 32, lane),
                 load_B_f32(Wmix, CS_, kt * 32, 96, cn0, lane, 1.f), acc);
#pragma unroll
  for (int kt = 0; kt < 2; ++kt)
    acc = wmma16(load_A_f32(ss, CS_, n0, kt * 32, lane),
                 load_B_f32(Wself, CS_, kt * 32, CS_, cn0, lane, 1.f), acc);
  int c = cn0 + (lane & 15);
  float cn = c_noise[0];
  float g1 = sigm(skb[c] + cn * skw[c]);
  float g2 = sigm(skb[64 + c] + cn * skw[64 + c]);
  int mb = n0 + ((lane >> 4) << 3);
#pragma unroll
  for (int r = 0; r < 8; ++r) {
    size_t id = (size_t)(mb + r) * CS_ + c;
    s_h[id] = g1 * s_h[id] + g2 * acc[r];
  }
}

// new_v[i] = a_v[i]@Wmix_v + vv[i]@Wself_v; fused skip-gate into v_h
__global__ void k_node_v(const float* __restrict__ a_v, const float* __restrict__ vvB,
                         const float* __restrict__ Wmix, const float* __restrict__ Wself,
                         const float* __restrict__ skw, const float* __restrict__ skb,
                         const float* __restrict__ c_noise, float* __restrict__ v_h) {
  int wave = (blockIdx.x * 256 + threadIdx.x) >> 5;
  int lane = threadIdx.x & 31;
  int tile = wave / 6, rem = wave % 6, pl = rem >> 1, ct = rem & 1;
  int n0 = tile * 16, cn0 = ct * 16;
  const float* Ab = a_v + (size_t)pl * N_ * 128;
  v8f acc = zero8();
#pragma unroll
  for (int kt = 0; kt < 4; ++kt)
    acc = wmma16(load_A_f32(Ab, 128, n0, kt * 32, lane),
                 load_B_f32(Wmix, CV_, kt * 32, 128, cn0, lane, 1.f), acc);
  acc = wmma16(load_A_f32(vvB + (size_t)pl * N_ * CV_, CV_, n0, 0, lane),
               load_B_f32(Wself, CV_, 0, CV_, cn0, lane, 1.f), acc);
  int cd = cn0 + (lane & 15);
  float cn = c_noise[0];
  float g3 = sigm(skb[128 + cd] + cn * skw[128 + cd]);
  float g4 = sigm(skb[160 + cd] + cn * skw[160 + cd]);
  float* Vp = v_h + (size_t)pl * N_ * CV_;
  int mb = n0 + ((lane >> 4) << 3);
#pragma unroll
  for (int r = 0; r < 8; ++r) {
    size_t id = (size_t)(mb + r) * CV_ + cd;
    Vp[id] = g3 * Vp[id] + g4 * acc[r];
  }
}

// out[n][i] = gain * sum_c v_h[i][n][c] * w_out[c]
__global__ void k_out(const float* __restrict__ v_h, const float* __restrict__ w_out,
                      const float* __restrict__ gain, float* __restrict__ out) {
  int idx = blockIdx.x * 256 + threadIdx.x;   // N*3 threads
  int n = idx / 3, i = idx % 3;
  const float* vp = v_h + ((size_t)i * N_ + n) * CV_;
  float acc = 0.f;
#pragma unroll
  for (int c = 0; c < CV_; ++c) acc += vp[c] * w_out[c];
  out[(size_t)n * 3 + i] = acc * gain[0];
}

// ---------------------------------------------------------------------------
// Host launcher
// ---------------------------------------------------------------------------
extern "C" void kernel_launch(void* const* d_in, const int* in_sizes, int n_in,
                              void* d_out, int out_size, void* d_ws, size_t ws_size,
                              hipStream_t stream) {
  (void)in_sizes; (void)n_in; (void)out_size; (void)ws_size;
  const float* pos      = (const float*)d_in[0];
  const float* c_noise  = (const float*)d_in[1];
  const int*   types    = (const int*)d_in[2];
  const int*   rad      = (const int*)d_in[3];
  const int*   bnd      = (const int*)d_in[4];
  const float* atom_tab = (const float*)d_in[5];
  const float* bond_tab = (const float*)d_in[6];
  const float* ns0_w    = (const float*)d_in[7];
  const float* We0      = (const float*)d_in[8];
  const float* We1      = (const float*)d_in[9];
  const float* Wself0   = (const float*)d_in[10];
  const float* Ws0      = (const float*)d_in[11];
  const float* Wv0      = (const float*)d_in[12];
  const float* ns_w     = (const float*)d_in[13];
  const float* We_ss    = (const float*)d_in[14];
  const float* We_vs    = (const float*)d_in[15];
  const float* We_sv    = (const float*)d_in[16];
  const float* We_vv    = (const float*)d_in[17];
  const float* We_vx    = (const float*)d_in[18];
  const float* Wmix_s   = (const float*)d_in[19];
  const float* Wmix_v   = (const float*)d_in[20];
  const float* Wself_s  = (const float*)d_in[21];
  const float* Wself_v  = (const float*)d_in[22];
  const float* skip_w   = (const float*)d_in[23];
  const float* skip_b   = (const float*)d_in[24];
  const float* w_out    = (const float*)d_in[25];
  const float* gain     = (const float*)d_in[26];

  char* ws = (char*)d_ws;
  size_t off = 0;
  auto carve = [&](size_t bytes) -> void* {
    void* p = ws + off;
    off += (bytes + 255) & ~(size_t)255;
    return p;
  };
  int*       srcA = (int*)carve((size_t)EE_ * 4);
  int*       dstA = (int*)carve((size_t)EE_ * 4);
  float*     Y1   = (float*)carve((size_t)EE_ * 3 * 4);
  _Float16*  eah  = (_Float16*)carve((size_t)EE_ * EA_ * 2);
  float*     sF   = (float*)carve((size_t)N_ * CS_ * 4);
  float*     s_h  = (float*)carve((size_t)N_ * CS_ * 4);
  float*     v_h  = (float*)carve((size_t)3 * N_ * CV_ * 4);
  float*     ssB  = (float*)carve((size_t)N_ * CS_ * 4);
  float*     vvB  = (float*)carve((size_t)3 * N_ * CV_ * 4);
  float*     a_s  = (float*)carve((size_t)N_ * 96 * 4);
  float*     a_v  = (float*)carve((size_t)3 * N_ * 128 * 4);
  _Float16*  pW   = (_Float16*)carve((size_t)36 * TILE_H * 2);
  _Float16*  pWe0 = pW;                     /* 4 tiles           */
  _Float16*  pWe1 = pW + 4  * TILE_H;       /* 4 tiles           */
  _Float16*  pWss = pW + 8  * TILE_H;       /* 2 layers x 4      */
  _Float16*  pWsv = pW + 16 * TILE_H;       /* 2 layers x 4      */
  _Float16*  pWvs = pW + 24 * TILE_H;       /* 2 layers x 2      */
  _Float16*  pWvv = pW + 28 * TILE_H;       /* 2 layers x 2      */
  _Float16*  pWvx = pW + 32 * TILE_H;       /* 2 layers x 2      */

  // Phase A: pack edge weights (WMMA B layout, 1/NORM folded), edge geom, embed
  k_packB<<<1, 128, 0, stream>>>(We0,   CS_, 4, 1, INV_NORM, pWe0);
  k_packB<<<1, 128, 0, stream>>>(We1,   CS_, 4, 1, INV_NORM, pWe1);
  k_packB<<<2, 128, 0, stream>>>(We_ss, CS_, 4, 2, INV_NORM, pWss);
  k_packB<<<2, 128, 0, stream>>>(We_sv, CS_, 4, 2, INV_NORM, pWsv);
  k_packB<<<1, 128, 0, stream>>>(We_vs, CV_, 2, 2, INV_NORM, pWvs);
  k_packB<<<1, 128, 0, stream>>>(We_vv, CV_, 2, 2, INV_NORM, pWvv);
  k_packB<<<1, 128, 0, stream>>>(We_vx, CV_, 2, 2, INV_NORM, pWvx);
  k_edge_geom<<<EE_ / 256, 256, 0, stream>>>(pos, rad, bnd, bond_tab, srcA, dstA, Y1, eah);
  k_embed<<<(N_ * CS_) / 256, 256, 0, stream>>>(atom_tab, types, ns0_w, c_noise, sF);

  // Layer-0 conv + node update
  k_zero<<<(N_ * 96) / 256, 256, 0, stream>>>(a_s, N_ * 96);
  k_zero<<<(3 * N_ * 128) / 256, 256, 0, stream>>>(a_v, 3 * N_ * 128);
  k_conv0<<<(EE_ / 16) / 8, 256, 0, stream>>>(eah, srcA, dstA, Y1, sF, pWe0, pWe1, a_s, a_v);
  k_node0_s<<<(1024 * 4) / 8, 256, 0, stream>>>(a_s, sF, Ws0, Wself0, s_h);
  k_node0_v<<<(1024 * 6) / 8, 256, 0, stream>>>(a_v, Wv0, v_h);

  // Interaction layers
  for (int l = 0; l < 2; ++l) {
    k_scale<<<(N_ * 96) / 256, 256, 0, stream>>>(s_h, v_h, ns_w + l * 96, c_noise, ssB, vvB);
    k_zero<<<(N_ * 96) / 256, 256, 0, stream>>>(a_s, N_ * 96);
    k_zero<<<(3 * N_ * 128) / 256, 256, 0, stream>>>(a_v, 3 * N_ * 128);
    k_conv<<<(EE_ / 16) / 8, 256, 0, stream>>>(
        eah, srcA, dstA, Y1, ssB, vvB,
        pWss + l * 4 * TILE_H, pWvs + l * 2 * TILE_H, pWsv + l * 4 * TILE_H,
        pWvv + l * 2 * TILE_H, pWvx + l * 2 * TILE_H, a_s, a_v);
    k_node_s<<<(1024 * 4) / 8, 256, 0, stream>>>(
        a_s, ssB, Wmix_s + l * 96 * CS_, Wself_s + l * CS_ * CS_,
        skip_w + l * 192, skip_b + l * 192, c_noise, s_h);
    k_node_v<<<(1024 * 6) / 8, 256, 0, stream>>>(
        a_v, vvB, Wmix_v + l * 128 * CV_, Wself_v + l * CV_ * CV_,
        skip_w + l * 192, skip_b + l * 192, c_noise, v_h);
  }

  k_out<<<(N_ * 3) / 256, 256, 0, stream>>>(v_h, w_out, gain, (float*)d_out);
}
